// TrueGroupedQueryAttention_84207128805949
// MI455X (gfx1250) — matrix operand order repaired and checked by reading, not compiled, not run on hardware
//
#include <hip/hip_runtime.h>

typedef unsigned short u16;
typedef unsigned int   u32;
typedef __attribute__((ext_vector_type(16))) __bf16 v16bf;
typedef __attribute__((ext_vector_type(8)))  float  v8f;

union Frag {
  v16bf v;
  u32   u[8];
  u16   h[16];
};

static __device__ __forceinline__ u16 f2bf(float f) {
  u32 u = __float_as_uint(f);
  u32 r = (u + 0x7FFFu + ((u >> 16) & 1u)) >> 16;  // round-to-nearest-even
  return (u16)r;
}
static __device__ __forceinline__ float bf2f(u16 h) {
  return __uint_as_float(((u32)h) << 16);
}
static __device__ __forceinline__ v8f wmma_bf16(const Frag& a, const Frag& b, v8f c) {
  // emits v_wmma_f32_16x16x32_bf16
  return __builtin_amdgcn_wmma_f32_16x16x32_bf16(false, a.v, false, b.v, (short)0, c,
                                                 false, false);
}

// ---------------------------------------------------------------------------
// f32 -> bf16 bulk convert
// ---------------------------------------------------------------------------
__global__ void f2bf_kernel(const float* __restrict__ in, u16* __restrict__ out, size_t n) {
  size_t i = (size_t)blockIdx.x * blockDim.x + threadIdx.x;
  if (i < n) out[i] = f2bf(in[i]);
}

// ---------------------------------------------------------------------------
// Tiled WMMA GEMM: C[M,N] = A[M,K] * W[K,N] + bias
// A,W bf16 row-major; block = 128 threads (4 waves); tile 128x64, BK=32.
// Wave w computes rows [w*32, w*32+32) x 64 cols: 2x4 WMMA accumulators,
// B-fragments reused across the two 16-row sub-tiles.
// ---------------------------------------------------------------------------
template <bool OUT_F32>
__global__ __launch_bounds__(128) void gemm_bf16(
    const u16* __restrict__ A, const u16* __restrict__ W,
    const float* __restrict__ bias, void* __restrict__ Cout,
    int M, int N, int K) {
  __shared__ u16 lA[128][34];  // A tile (padded rows: kill bank conflicts)
  __shared__ u16 lWt[64][34];  // W tile transposed: [n][k]

  const int tid     = threadIdx.x;
  const int lane    = tid & 31;
  const int w       = tid >> 5;
  const int lane15  = lane & 15;
  const int halfsel = (lane >> 4) & 1;
  const int bm = blockIdx.x * 128;
  const int bn = blockIdx.y * 64;

  v8f acc[2][4] = {};

  for (int k0 = 0; k0 < K; k0 += 32) {
    // ---- stage A tile: 128x32 bf16 = 2048 u32, 16 per thread ----
#pragma unroll
    for (int r = 0; r < 16; ++r) {
      int idx  = tid + r * 128;  // 0..2047
      int row  = idx >> 4;       // 16 u32 per row
      int col2 = idx & 15;
      *(u32*)&lA[row][2 * col2] =
          *(const u32*)(A + (size_t)(bm + row) * K + k0 + 2 * col2);
    }
    // ---- stage W tile transposed: W[k0+k][bn+n] -> lWt[n][k] ----
#pragma unroll
    for (int r = 0; r < 16; ++r) {
      int idx = tid + r * 128;  // 0..2047
      int kk  = idx >> 6;       // 0..31
      int nn  = idx & 63;
      lWt[nn][kk] = W[(size_t)(k0 + kk) * N + bn + nn];
    }
    // prefetch next K-tile into cache (global_prefetch_b8)
    if (k0 + 32 < K) {
      __builtin_prefetch(A + (size_t)(bm + (tid >> 4) * 8) * K + k0 + 32, 0, 1);
      __builtin_prefetch(W + (size_t)(k0 + 32 + (tid >> 4)) * N + bn + (tid & 15) * 4, 0, 1);
    }
    __syncthreads();

    // ---- A fragments (16x32, A layout) for the two row sub-tiles ----
    Frag a0, a1;
    int r0 = w * 32 + lane15;
#pragma unroll
    for (int i = 0; i < 8; ++i) {
      int kk = (i >> 2) * 16 + halfsel * 8 + (i & 3) * 2;
      a0.u[i] = *(const u32*)&lA[r0][kk];
      a1.u[i] = *(const u32*)&lA[r0 + 16][kk];
    }
    // ---- 4 N-tiles: B fragment (32x16, B layout) reused for both rows ----
#pragma unroll
    for (int t = 0; t < 4; ++t) {
      Frag b;
      int ncol = t * 16 + lane15;
#pragma unroll
      for (int i = 0; i < 8; ++i) {
        int kk = (i >> 2) * 16 + halfsel * 8 + (i & 3) * 2;
        b.u[i] = *(const u32*)&lWt[ncol][kk];
      }
      acc[0][t] = wmma_bf16(a0, b, acc[0][t]);
      acc[1][t] = wmma_bf16(a1, b, acc[1][t]);
    }
    __syncthreads();
  }

  // ---- epilogue: C layout row = v + 8*halfsel, col = lane15 ----
#pragma unroll
  for (int s = 0; s < 2; ++s) {
    int rowBase = bm + w * 32 + s * 16;
#pragma unroll
    for (int t = 0; t < 4; ++t) {
      int col = bn + t * 16 + lane15;
      float bv = bias[col];
#pragma unroll
      for (int v = 0; v < 8; ++v) {
        int row = rowBase + v + 8 * halfsel;
        float val = acc[s][t][v] + bv;
        if constexpr (OUT_F32)
          ((float*)Cout)[(size_t)row * N + col] = val;
        else
          ((u16*)Cout)[(size_t)row * N + col] = f2bf(val);
      }
    }
  }
}

// ---------------------------------------------------------------------------
// RoPE in place on bf16 tensor shaped (B, S, H, 64). One thread per (pos, j<32).
// outscale folds the attention 1/sqrt(d) into Q (scaling commutes with RoPE).
// ---------------------------------------------------------------------------
__global__ void rope_kernel(u16* __restrict__ x, int S, int H, float outscale,
                            size_t total) {
  size_t idx = (size_t)blockIdx.x * blockDim.x + threadIdx.x;
  if (idx >= total) return;
  int j = (int)(idx & 31);
  size_t rest = idx >> 5;  // ((b*S + s)*H + h)
  int s = (int)((rest / H) % S);
  size_t base = rest * 64;
  float inv = __expf(-(float)j * (__logf(10000.0f) / 32.0f));
  float ang = (float)s * inv;
  float sn, c;
  __sincosf(ang, &sn, &c);
  float x1 = bf2f(x[base + j]);
  float x2 = bf2f(x[base + j + 32]);
  x[base + j]      = f2bf((x1 * c - x2 * sn) * outscale);
  x[base + j + 32] = f2bf((x2 * c + x1 * sn) * outscale);
}

// ---------------------------------------------------------------------------
// Causal flash attention.
// q: (B,S,4,64) bf16 (post-RoPE, pre-scaled by 1/8), k,v: (B,S,16,64) bf16,
// out: (B,S,16,64) bf16.
// Block = 128 threads (4 waves) owns (b, h, 64-query tile); wave w owns a
// 16-query sub-tile. Uniform j-loop across the block: K/V 32x64 tiles staged
// in LDS once per block (V transposed via packed u32 stores). Softmax row
// sums come from an extra WMMA against an all-ones B fragment; only diagonal
// blocks take the masked path.
// ---------------------------------------------------------------------------
__global__ __launch_bounds__(128) void attn_kernel(
    const u16* __restrict__ qb, const u16* __restrict__ kb,
    const u16* __restrict__ vb, u16* __restrict__ ob, int B, int S) {
  __shared__ u16 kT[32][66];      // [key][d], padded
  __shared__ u16 vT[64][34];      // [d][key], padded
  __shared__ u16 pL[4][16 * 34];  // per-wave P staging, padded rows

  const int tid     = threadIdx.x;
  const int lane    = tid & 31;
  const int w       = tid >> 5;
  const int lane15  = lane & 15;
  const int halfsel = (lane >> 4) & 1;

  const int nqb = S / 64;
  int bid  = blockIdx.x;
  int qblk = bid % nqb;
  int t2   = bid / nqb;
  int h    = t2 % 16;
  int b    = t2 / 16;
  int g    = h >> 2;  // query group (jnp.repeat semantics)
  int q0   = qblk * 64 + w * 16;

  // ---- Q fragments (A layout), d split 0..31 / 32..63 ----
  Frag aq0, aq1;
  {
    size_t qbase = ((size_t)(b * S + q0 + lane15) * 4 + g) * 64;
#pragma unroll
    for (int i = 0; i < 8; ++i) {
      int kk = (i >> 2) * 16 + halfsel * 8 + (i & 3) * 2;
      aq0.u[i] = *(const u32*)(qb + qbase + kk);
      aq1.u[i] = *(const u32*)(qb + qbase + 32 + kk);
    }
  }
  Frag ones;  // all-ones B fragment: row-sum via WMMA
#pragma unroll
  for (int i = 0; i < 8; ++i) ones.u[i] = 0x3F803F80u;

  float m_run[8];
  int   myq[8];
#pragma unroll
  for (int v = 0; v < 8; ++v) {
    m_run[v] = -3.0e38f;
    myq[v]   = q0 + v + 8 * halfsel;
  }
  v8f acc[4] = {};
  v8f accl   = {};  // softmax denominators (all 16 columns identical)

  const int jEnd = qblk * 64 + 63;  // uniform across block -> __syncthreads legal
  for (int j0 = 0; j0 <= jEnd; j0 += 32) {
    // ---- stage K tile row-major: 1024 u32, 8 per thread ----
#pragma unroll
    for (int r = 0; r < 8; ++r) {
      int idx = tid + r * 128;  // 32 u32 per key-row
      int key = idx >> 5;
      int c2  = idx & 31;
      size_t rowbase = ((size_t)(b * S + j0 + key) * 16 + h) * 64;
      *(u32*)&kT[key][2 * c2] = *(const u32*)(kb + rowbase + 2 * c2);
    }
    // ---- stage V transposed with packed u32 stores: each thread takes a
    //      (key-pair, d-pair) unit -> 2 global u32 loads, 2 LDS u32 stores ----
#pragma unroll
    for (int r = 0; r < 4; ++r) {
      int idx = tid + r * 128;  // 0..511 units
      int kp  = idx >> 5;       // key pair 0..15
      int c2  = idx & 31;       // d pair 0..31
      size_t r0 = ((size_t)(b * S + j0 + 2 * kp) * 16 + h) * 64 + 2 * c2;
      u32 a = *(const u32*)(vb + r0);            // key 2kp   : d 2c2, 2c2+1
      u32 c = *(const u32*)(vb + r0 + 16 * 64);  // key 2kp+1 : d 2c2, 2c2+1
      *(u32*)&vT[2 * c2][2 * kp]     = (a & 0xFFFFu) | (c << 16);
      *(u32*)&vT[2 * c2 + 1][2 * kp] = (a >> 16) | (c & 0xFFFF0000u);
    }
    // prefetch next key-block (global_prefetch_b8)
    if (j0 + 32 <= jEnd) {
      size_t nb = ((size_t)(b * S + j0 + 32 + (tid >> 2)) * 16 + h) * 64;
      __builtin_prefetch(kb + nb, 0, 1);
      __builtin_prefetch(vb + nb, 0, 1);
    }
    __syncthreads();

    // ---- scores: two 16-key tiles, 2 WMMAs each (d 0..31 / 32..63) ----
    v8f sc[2];
#pragma unroll
    for (int t = 0; t < 2; ++t) {
      Frag bk0, bk1;  // B layout: lane = key column, K dim = d
      int krow = t * 16 + lane15;
#pragma unroll
      for (int i = 0; i < 8; ++i) {
        int kk = (i >> 2) * 16 + halfsel * 8 + (i & 3) * 2;
        bk0.u[i] = *(const u32*)&kT[krow][kk];
        bk1.u[i] = *(const u32*)&kT[krow][32 + kk];
      }
      v8f z = {};
      z = wmma_bf16(aq0, bk0, z);
      z = wmma_bf16(aq1, bk1, z);
      sc[t] = z;
    }

    // ---- online softmax (scale already folded into Q) ----
    const bool needMask = (j0 + 31 > q0);  // wave-uniform -> scalar branch
#pragma unroll
    for (int v = 0; v < 8; ++v) {
      float s0 = sc[0][v];
      float s1 = sc[1][v];
      if (needMask) {
        int qi = myq[v];
        s0 = (j0 + lane15 <= qi) ? s0 : -1.0e30f;
        s1 = (j0 + 16 + lane15 <= qi) ? s1 : -1.0e30f;
      }
      float rm = fmaxf(s0, s1);
#pragma unroll
      for (int d = 1; d < 16; d <<= 1) rm = fmaxf(rm, __shfl_xor(rm, d, 32));
      float m_new = fmaxf(m_run[v], rm);
      float p0   = __expf(s0 - m_new);
      float p1   = __expf(s1 - m_new);
      float corr = __expf(m_run[v] - m_new);
      m_run[v] = m_new;
      acc[0][v] *= corr; acc[1][v] *= corr;
      acc[2][v] *= corr; acc[3][v] *= corr;
      accl[v]   *= corr;
      int row = v + 8 * halfsel;
      pL[w][row * 34 + lane15]      = f2bf(p0);
      pL[w][row * 34 + 16 + lane15] = f2bf(p1);
    }

    // same-wave LDS RAW: P written above, read below
    asm volatile("s_wait_dscnt 0" ::: "memory");

    // ---- P fragment (A layout, 16 queries x 32 keys) ----
    Frag pa;
#pragma unroll
    for (int i = 0; i < 8; ++i) {
      int kk = (i >> 2) * 16 + halfsel * 8 + (i & 3) * 2;
      pa.u[i] = *(const u32*)&pL[w][lane15 * 34 + kk];
    }

    // ---- V fragments (B layout: lane = d column, K dim = keys) + row sums ----
#pragma unroll
    for (int f = 0; f < 4; ++f) {
      Frag vf;
      int drow = f * 16 + lane15;
#pragma unroll
      for (int i = 0; i < 8; ++i) {
        int kk = (i >> 2) * 16 + halfsel * 8 + (i & 3) * 2;  // key pair, contiguous in vT
        vf.u[i] = *(const u32*)&vT[drow][kk];
      }
      acc[f] = wmma_bf16(pa, vf, acc[f]);
    }
    accl = wmma_bf16(pa, ones, accl);  // l += P . 1  (replaces sum butterfly)
    __syncthreads();  // kT/vT reused next iteration
  }

  // ---- normalize and store (B,S,16,64) bf16 ----
#pragma unroll
  for (int v = 0; v < 8; ++v) {
    float inv = 1.0f / accl[v];
    int q = q0 + v + 8 * halfsel;
    size_t obase = ((size_t)(b * S + q) * 16 + h) * 64;
#pragma unroll
    for (int f = 0; f < 4; ++f) {
      ob[obase + f * 16 + lane15] = f2bf(acc[f][v] * inv);
    }
  }
}

// ---------------------------------------------------------------------------
// Host launcher
// ---------------------------------------------------------------------------
extern "C" void kernel_launch(void* const* d_in, const int* in_sizes, int n_in,
                              void* d_out, int out_size, void* d_ws, size_t ws_size,
                              hipStream_t stream) {
  (void)in_sizes; (void)n_in; (void)out_size; (void)ws_size;
  const int B = 2, S = 2048, D = 1024, M = B * S;

  const float* x  = (const float*)d_in[0];
  const float* wq = (const float*)d_in[1];
  const float* bq = (const float*)d_in[2];
  const float* wk = (const float*)d_in[3];
  const float* bk = (const float*)d_in[4];
  const float* wv = (const float*)d_in[5];
  const float* bv = (const float*)d_in[6];
  const float* wo = (const float*)d_in[7];
  const float* bo = (const float*)d_in[8];

  char* ws = (char*)d_ws;
  size_t off = 0;
  auto alloc = [&](size_t bytes) -> void* {
    void* p = ws + off;
    off += (bytes + 255) & ~(size_t)255;
    return p;
  };
  u16* xb  = (u16*)alloc((size_t)M * D * 2);
  u16* wqb = (u16*)alloc((size_t)D * 256 * 2);
  u16* wkb = (u16*)alloc((size_t)D * D * 2);
  u16* wvb = (u16*)alloc((size_t)D * D * 2);
  u16* wob = (u16*)alloc((size_t)D * D * 2);
  u16* q_  = (u16*)alloc((size_t)M * 256 * 2);
  u16* k_  = (u16*)alloc((size_t)M * D * 2);
  u16* v_  = (u16*)alloc((size_t)M * D * 2);
  u16* o_  = (u16*)alloc((size_t)M * D * 2);

  auto cvt = [&](const float* src, u16* dst, size_t n) {
    f2bf_kernel<<<(unsigned)((n + 255) / 256), 256, 0, stream>>>(src, dst, n);
  };
  cvt(x,  xb,  (size_t)M * D);
  cvt(wq, wqb, (size_t)D * 256);
  cvt(wk, wkb, (size_t)D * D);
  cvt(wv, wvb, (size_t)D * D);
  cvt(wo, wob, (size_t)D * D);

  dim3 blk(128);
  gemm_bf16<false><<<dim3(M / 128, 256 / 64), blk, 0, stream>>>(xb, wqb, bq, q_, M, 256, D);
  gemm_bf16<false><<<dim3(M / 128, D / 64),   blk, 0, stream>>>(xb, wkb, bk, k_, M, D, D);
  gemm_bf16<false><<<dim3(M / 128, D / 64),   blk, 0, stream>>>(xb, wvb, bv, v_, M, D, D);

  rope_kernel<<<(B * S * 4  * 32) / 256, 256, 0, stream>>>(q_, S, 4, 0.125f,
                                                           (size_t)B * S * 4 * 32);
  rope_kernel<<<(B * S * 16 * 32) / 256, 256, 0, stream>>>(k_, S, 16, 1.0f,
                                                           (size_t)B * S * 16 * 32);

  attn_kernel<<<B * 16 * (S / 64), 128, 0, stream>>>(q_, k_, v_, o_, B, S);

  gemm_bf16<true><<<dim3(M / 128, D / 64), blk, 0, stream>>>(o_, wob, bo, d_out, M, D, D);
}
